// IL_GAT_81372450390811
// MI455X (gfx1250) — matrix operand order repaired and checked by reading, not compiled, not run on hardware
//
#include <hip/hip_runtime.h>
#include <hip/hip_bf16.h>
#include <math.h>

typedef __attribute__((ext_vector_type(16))) __bf16 v16bf;
typedef __attribute__((ext_vector_type(8)))  __bf16 v8bf;
typedef __attribute__((ext_vector_type(8)))  float  v8f;

#define N_NODES   20000
#define N_EDGES   320000
#define N_TOT_E   (N_EDGES + N_NODES)   // edges + self loops
#define N_GRAPHS  200
#define NEG_SLOPE 0.2f
#define BN_EPS    1e-5f

// ---------------------------------------------------------------- helpers
static __device__ __forceinline__ unsigned f2bf_bits(float f) {
  union { float f; unsigned u; } v; v.f = f;
  return (v.u + 0x7FFFu + ((v.u >> 16) & 1u)) >> 16;   // RNE to bf16 bits
}

// fp32 -> bf16, two elements per thread (packed dword stores)
__global__ __launch_bounds__(256)
void cvt_bf16(const float* __restrict__ in, unsigned int* __restrict__ out, size_t n2) {
  size_t i = (size_t)blockIdx.x * blockDim.x + threadIdx.x;
  if (i >= n2) return;
  unsigned lo = f2bf_bits(in[2 * i]);
  unsigned hi = f2bf_bits(in[2 * i + 1]);
  out[i] = lo | (hi << 16);
}

// Pack W[K x Nd] (fp32 row-major) into WMMA-native bf16 B fragments:
// Bpack[ct][kt][lane][i] = { W[kb][col], W[kb+1][col] },
//   kb = kt*32 + 2*(i&3) + 8*(lane>>4) + 16*(i>>2), col = ct*16 + (lane&15)
__global__ __launch_bounds__(256)
void pack_weight(const float* __restrict__ W, unsigned int* __restrict__ out,
                 int K, int Nd) {
  int idx = blockIdx.x * blockDim.x + threadIdx.x;
  int kT = K >> 5;
  int total = (Nd >> 4) * kT * 256;
  if (idx >= total) return;
  int i    = idx & 7;
  int lane = (idx >> 3) & 31;
  int t    = idx >> 8;               // ct*kT + kt
  int kt   = t % kT, ct = t / kT;
  int half = lane >> 4, m16 = lane & 15;
  int kb   = (kt << 5) + ((i & 3) << 1) + (half << 3) + ((i >> 2) << 4);
  int col  = (ct << 4) + m16;
  unsigned lo = f2bf_bits(W[(size_t)kb * Nd + col]);
  unsigned hi = f2bf_bits(W[(size_t)(kb + 1) * Nd + col]);
  out[idx] = lo | (hi << 16);
}

// ---------------------------------------------------------------- GEMM (bf16 WMMA)
// C[N x Nd] = A[N x K] @ B[K x Nd]; A is bf16 row-major, B pre-packed.
// One wave -> 32x32 C tile (4 WMMAs/K-step); block = 4 waves -> 32 x 128.
__global__ __launch_bounds__(128)
void gemm_wmma32(const __bf16* __restrict__ Ab, const unsigned int* __restrict__ Bp,
                 float* __restrict__ C, int K, int Nd) {
  const int lane = threadIdx.x & 31;
  const int wave = threadIdx.x >> 5;
  const int half = lane >> 4;
  const int m16  = lane & 15;
  const int kT   = K >> 5;
  const int row0 = blockIdx.y << 5;
  const int col0 = (blockIdx.x * 4 + wave) << 5;

  const __bf16* ar0 = Ab + (size_t)(row0 + m16) * K + (half << 3);
  const __bf16* ar1 = ar0 + (size_t)16 * K;
  const unsigned int* bp0 = Bp + ((size_t)(col0 >> 4) * kT * 32 + lane) * 8;
  const unsigned int* bp1 = bp0 + (size_t)kT * 256;

  v8f acc00 = {}, acc01 = {}, acc10 = {}, acc11 = {};
  for (int kt = 0; kt < kT; ++kt) {
    const __bf16* pa0 = ar0 + (kt << 5);
    const __bf16* pa1 = ar1 + (kt << 5);
    v8bf a0lo = *(const v8bf*)(pa0);
    v8bf a0hi = *(const v8bf*)(pa0 + 16);
    v8bf a1lo = *(const v8bf*)(pa1);
    v8bf a1hi = *(const v8bf*)(pa1 + 16);
    v16bf a0 = __builtin_shufflevector(a0lo, a0hi,
        0,1,2,3,4,5,6,7,8,9,10,11,12,13,14,15);
    v16bf a1 = __builtin_shufflevector(a1lo, a1hi,
        0,1,2,3,4,5,6,7,8,9,10,11,12,13,14,15);
    v16bf b0 = *(const v16bf*)(bp0 + (size_t)kt * 256);
    v16bf b1 = *(const v16bf*)(bp1 + (size_t)kt * 256);
    acc00 = __builtin_amdgcn_wmma_f32_16x16x32_bf16(false, a0, false, b0, (short)0, acc00, false, false);
    acc01 = __builtin_amdgcn_wmma_f32_16x16x32_bf16(false, a0, false, b1, (short)0, acc01, false, false);
    acc10 = __builtin_amdgcn_wmma_f32_16x16x32_bf16(false, a1, false, b0, (short)0, acc10, false, false);
    acc11 = __builtin_amdgcn_wmma_f32_16x16x32_bf16(false, a1, false, b1, (short)0, acc11, false, false);
  }
  // C/D layout: VGPR r -> M = r + 8*half, N = lane&15
#pragma unroll
  for (int r = 0; r < 8; ++r) {
    int rowa = row0 + (half << 3) + r;
    int rowb = rowa + 16;
    C[(size_t)rowa * Nd + col0 + m16]      = acc00[r];
    C[(size_t)rowa * Nd + col0 + 16 + m16] = acc01[r];
    C[(size_t)rowb * Nd + col0 + m16]      = acc10[r];
    C[(size_t)rowb * Nd + col0 + 16 + m16] = acc11[r];
  }
}

// ---------------------------------------------------------------- edge phase
__global__ __launch_bounds__(256)
void edge_score(const float* __restrict__ xl, const float* __restrict__ xr,
                const int* __restrict__ srcA, const int* __restrict__ dstA,
                const float* __restrict__ att, float* __restrict__ e,
                float* __restrict__ mx, int D) {
  int eid = blockIdx.x * (blockDim.x >> 5) + (threadIdx.x >> 5);
  if (eid >= N_TOT_E) return;
  int lane = threadIdx.x & 31;
  int s, d;
  if (eid < N_EDGES) { s = srcA[eid]; d = dstA[eid]; }
  else               { s = d = eid - N_EDGES; }          // self loop
  const float* pl = xl + (size_t)s * D;
  const float* pr = xr + (size_t)d * D;
  float acc = 0.f;
  for (int f = lane; f < D; f += 32) {
    float v = pl[f] + pr[f];
    v = v > 0.f ? v : NEG_SLOPE * v;
    acc += att[f] * v;
  }
  for (int off = 16; off; off >>= 1) acc += __shfl_xor(acc, off, 32);
  if (lane == 0) {
    e[eid] = acc;
    // float atomic max via int punning; mx pre-filled with 0xFFFFFFFF
    if (acc >= 0.f) atomicMax((int*)(mx + d), __float_as_int(acc));
    else            atomicMin((unsigned int*)(mx + d), __float_as_uint(acc));
  }
}

__global__ __launch_bounds__(256)
void edge_exp(const float* __restrict__ e, const int* __restrict__ dstA,
              const float* __restrict__ mx, float* __restrict__ ex,
              float* __restrict__ denom) {
  int eid = blockIdx.x * blockDim.x + threadIdx.x;
  if (eid >= N_TOT_E) return;
  int d = (eid < N_EDGES) ? dstA[eid] : (eid - N_EDGES);
  float v = expf(e[eid] - mx[d]);
  ex[eid] = v;
  atomicAdd(denom + d, v);
}

__global__ __launch_bounds__(256)
void edge_aggregate(const float* __restrict__ xl,
                    const int* __restrict__ srcA, const int* __restrict__ dstA,
                    const float* __restrict__ ex, const float* __restrict__ denom,
                    float* __restrict__ out, int D) {
  int eid = blockIdx.x * (blockDim.x >> 5) + (threadIdx.x >> 5);
  if (eid >= N_TOT_E) return;
  int lane = threadIdx.x & 31;
  int s, d;
  if (eid < N_EDGES) { s = srcA[eid]; d = dstA[eid]; }
  else               { s = d = eid - N_EDGES; }
  float coeff = ex[eid] / denom[d];
  const float* pl = xl + (size_t)s * D;
  float* po = out + (size_t)d * D;
  for (int f = lane; f < D; f += 32) atomicAdd(po + f, coeff * pl[f]);
}

__global__ __launch_bounds__(256)
void bias_relu(float* __restrict__ h, const float* __restrict__ bias,
               int D, size_t total) {
  size_t i = (size_t)blockIdx.x * blockDim.x + threadIdx.x;
  if (i >= total) return;
  float v = h[i] + bias[i & (D - 1)];   // D is a power of two (512/1024)
  h[i] = v > 0.f ? v : 0.f;
}

// ---------------------------------------------------------------- readout + MLP
__global__ __launch_bounds__(256)
void find_last(const int* __restrict__ batch, int* __restrict__ lastidx) {
  int i = blockIdx.x * blockDim.x + threadIdx.x;
  if (i >= N_NODES) return;
  if (i == N_NODES - 1 || batch[i] != batch[i + 1]) lastidx[batch[i]] = i;
}

__global__ __launch_bounds__(256)
void build_g(const float* __restrict__ h, const int* __restrict__ lastidx,
             const float* __restrict__ cond, float* __restrict__ g) {
  int i = blockIdx.x * blockDim.x + threadIdx.x;
  if (i >= N_GRAPHS * 514) return;
  int gr = i / 514, c = i % 514;
  g[i] = (c < 512) ? h[(size_t)lastidx[gr] * 512 + c] : cond[gr * 2 + (c - 512)];
}

__global__ __launch_bounds__(256)
void mlp_bn_relu(const float* __restrict__ in, const float* __restrict__ W,
                 const float* __restrict__ b,
                 const float* __restrict__ bng, const float* __restrict__ bnb,
                 const float* __restrict__ bnm, const float* __restrict__ bnv,
                 float* __restrict__ out, int din, int dout) {
  int i = blockIdx.x * blockDim.x + threadIdx.x;
  if (i >= N_GRAPHS * dout) return;
  int gr = i / dout, c = i % dout;
  const float* row = in + (size_t)gr * din;
  float acc = b[c];
  for (int k = 0; k < din; ++k) acc += row[k] * W[(size_t)k * dout + c];
  acc = (acc - bnm[c]) * rsqrtf(bnv[c] + BN_EPS) * bng[c] + bnb[c];
  out[i] = acc > 0.f ? acc : 0.f;
}

__global__ __launch_bounds__(256)
void mlp_final(const float* __restrict__ in, const float* __restrict__ W,
               const float* __restrict__ b, float* __restrict__ out, int din) {
  int gr = blockIdx.x * blockDim.x + threadIdx.x;
  if (gr >= N_GRAPHS) return;
  const float* row = in + (size_t)gr * din;
  float acc = b[0];
  for (int k = 0; k < din; ++k) acc += row[k] * W[k];
  out[gr] = acc;
}

// ---------------------------------------------------------------- launch
extern "C" void kernel_launch(void* const* d_in, const int* in_sizes, int n_in,
                              void* d_out, int out_size, void* d_ws, size_t ws_size,
                              hipStream_t stream) {
  (void)in_sizes; (void)n_in; (void)out_size; (void)ws_size;
  const float* x    = (const float*)d_in[0];
  const int*   ei   = (const int*)d_in[1];
  const int*   bat  = (const int*)d_in[2];
  const float* cond = (const float*)d_in[3];
  const float* Wl[4]  = {(const float*)d_in[4],  (const float*)d_in[8],
                         (const float*)d_in[12], (const float*)d_in[16]};
  const float* Wr[4]  = {(const float*)d_in[5],  (const float*)d_in[9],
                         (const float*)d_in[13], (const float*)d_in[17]};
  const float* att[4] = {(const float*)d_in[6],  (const float*)d_in[10],
                         (const float*)d_in[14], (const float*)d_in[18]};
  const float* bia[4] = {(const float*)d_in[7],  (const float*)d_in[11],
                         (const float*)d_in[15], (const float*)d_in[19]};
  const float* W5a = (const float*)d_in[20]; const float* b5a = (const float*)d_in[21];
  const float* W5b = (const float*)d_in[22]; const float* b5b = (const float*)d_in[23];
  const float* W5c = (const float*)d_in[24]; const float* b5c = (const float*)d_in[25];
  const float* bn1g = (const float*)d_in[26]; const float* bn1b = (const float*)d_in[27];
  const float* bn1m = (const float*)d_in[28]; const float* bn1v = (const float*)d_in[29];
  const float* bn2g = (const float*)d_in[30]; const float* bn2b = (const float*)d_in[31];
  const float* bn2m = (const float*)d_in[32]; const float* bn2v = (const float*)d_in[33];

  const int* srcA = ei;
  const int* dstA = ei + N_EDGES;

  // workspace carve-up (256B aligned)
  char* ws = (char*)d_ws;
  size_t off = 0;
  auto carve = [&](size_t bytes) -> void* {
    void* p = ws + off; off += (bytes + 255) & ~(size_t)255; return p;
  };
  float* h     = (float*)carve((size_t)N_NODES * 1024 * 4);
  float* xl    = (float*)carve((size_t)N_NODES * 1024 * 4);
  float* xr    = (float*)carve((size_t)N_NODES * 1024 * 4);
  unsigned int* Ab  = (unsigned int*)carve((size_t)N_NODES * 1024 * 2); // bf16 activations
  unsigned int* BpL = (unsigned int*)carve((size_t)1024 * 1024 * 2);    // packed bf16 weights
  unsigned int* BpR = (unsigned int*)carve((size_t)1024 * 1024 * 2);
  float* ebuf  = (float*)carve((size_t)N_TOT_E * 4);
  float* exbuf = (float*)carve((size_t)N_TOT_E * 4);
  float* mx    = (float*)carve((size_t)N_NODES * 4);
  float* den   = (float*)carve((size_t)N_NODES * 4);
  int*   lastI = (int*)  carve((size_t)N_GRAPHS * 4);
  float* gmat  = (float*)carve((size_t)N_GRAPHS * 514 * 4);
  float* t1    = (float*)carve((size_t)N_GRAPHS * 1024 * 4);
  float* t2    = (float*)carve((size_t)N_GRAPHS * 512 * 4);

  const int dinA[4]  = {64, 512, 1024, 1024};
  const int doutA[4] = {512, 1024, 1024, 512};
  const int eWaves = (N_TOT_E * 32 + 255) / 256;

  for (int li = 0; li < 4; ++li) {
    const int din = dinA[li], dout = doutA[li];
    const float* Aptr = (li == 0) ? x : h;

    // stage: activations -> bf16, weights -> packed WMMA fragments
    size_t n2 = (size_t)N_NODES * din / 2;
    cvt_bf16<<<(int)((n2 + 255) / 256), 256, 0, stream>>>(Aptr, Ab, n2);
    int wtot = (dout >> 4) * (din >> 5) * 256;
    pack_weight<<<(wtot + 255) / 256, 256, 0, stream>>>(Wl[li], BpL, din, dout);
    pack_weight<<<(wtot + 255) / 256, 256, 0, stream>>>(Wr[li], BpR, din, dout);

    dim3 gg(dout / 128, N_NODES / 32);   // 32x128 C per block (4 waves, 32x32 each)
    gemm_wmma32<<<gg, 128, 0, stream>>>((const __bf16*)Ab, BpL, xl, din, dout);
    gemm_wmma32<<<gg, 128, 0, stream>>>((const __bf16*)Ab, BpR, xr, din, dout);

    hipMemsetAsync(mx,  0xFF, (size_t)N_NODES * 4, stream);
    hipMemsetAsync(den, 0,    (size_t)N_NODES * 4, stream);
    hipMemsetAsync(h,   0,    (size_t)N_NODES * dout * 4, stream);

    edge_score<<<eWaves, 256, 0, stream>>>(xl, xr, srcA, dstA, att[li], ebuf, mx, dout);
    edge_exp<<<(N_TOT_E + 255) / 256, 256, 0, stream>>>(ebuf, dstA, mx, exbuf, den);
    edge_aggregate<<<eWaves, 256, 0, stream>>>(xl, srcA, dstA, exbuf, den, h, dout);

    size_t tot = (size_t)N_NODES * dout;
    bias_relu<<<(int)((tot + 255) / 256), 256, 0, stream>>>(h, bia[li], dout, tot);
  }

  find_last<<<(N_NODES + 255) / 256, 256, 0, stream>>>(bat, lastI);
  build_g<<<(N_GRAPHS * 514 + 255) / 256, 256, 0, stream>>>(h, lastI, cond, gmat);
  mlp_bn_relu<<<(N_GRAPHS * 1024 + 255) / 256, 256, 0, stream>>>(
      gmat, W5a, b5a, bn1g, bn1b, bn1m, bn1v, t1, 514, 1024);
  mlp_bn_relu<<<(N_GRAPHS * 512 + 255) / 256, 256, 0, stream>>>(
      t1, W5b, b5b, bn2g, bn2b, bn2m, bn2v, t2, 1024, 512);
  mlp_final<<<(N_GRAPHS + 255) / 256, 256, 0, stream>>>(t2, W5c, b5c, (float*)d_out, 512);
}
